// GOA_TF_21612275433493
// MI455X (gfx1250) — matrix operand order repaired and checked by reading, not compiled, not run on hardware
//
#include <hip/hip_runtime.h>
#include <math.h>

typedef __attribute__((ext_vector_type(16))) _Float16 v16h;
typedef __attribute__((ext_vector_type(8)))  float    v8f;

union HF { v16h v; unsigned int u[8]; };

__device__ __forceinline__ float wave_sum32(float v) {
#pragma unroll
  for (int m = 16; m >= 1; m >>= 1) v += __shfl_xor(v, m, 32);
  return v;
}
__device__ __forceinline__ float half_sum16(float v) {
#pragma unroll
  for (int m = 8; m >= 1; m >>= 1) v += __shfl_xor(v, m, 32);
  return v;
}
__device__ __forceinline__ float half_max16(float v) {
#pragma unroll
  for (int m = 8; m >= 1; m >>= 1) v = fmaxf(v, __shfl_xor(v, m, 32));
  return v;
}

// ---------------------------------------------------------------------------
// WMMA GEMM: C[N,M] = act(A[N,K] @ W[K,M] + bias) (+ res).
// Tile 128x128, K-step 32, 8 waves (4 row-groups x 2 col-groups, 32x64 each).
// Register double-buffering: next k-tile global loads overlap WMMA compute.
// ---------------------------------------------------------------------------
__global__ __launch_bounds__(256) void gemm_wmma_kernel(
    const float* __restrict__ A, const float* __restrict__ W,
    const float* __restrict__ bias, const float* __restrict__ res,
    float* __restrict__ C, int K, int M, int act) {
  constexpr int AST = 34, WST = 34;
  __shared__ __align__(16) _Float16 As[128 * AST];
  __shared__ __align__(16) _Float16 Ws[128 * WST];
  const int tid = threadIdx.x;
  const int lane = tid & 31;
  const int wv = tid >> 5;
  const int wr = wv & 3;          // row group 0..3 (32 rows each)
  const int wcg = wv >> 2;        // col group 0..1 (64 cols each)
  const int row0 = blockIdx.y * 128;
  const int col0 = blockIdx.x * 128;
  const int l15 = lane & 15;
  const int hlf = lane >> 4;

  v8f acc[2][4];
  {
    v8f vzero = {0.f, 0.f, 0.f, 0.f, 0.f, 0.f, 0.f, 0.f};
#pragma unroll
    for (int rt = 0; rt < 2; rt++)
#pragma unroll
      for (int ct = 0; ct < 4; ct++) acc[rt][ct] = vzero;
  }

  const int tr = tid >> 1;               // A row per thread
  const int tcb = (tid & 1) * 16;        // A col base
  const int wkb = (tid >> 4) * 2;        // W k base (2 rows of k per thread)
  const int wnb = (tid & 15) * 8;        // W n base (8 cols per thread)

  float4 aReg[4];
  float wReg[16];

#define GEMM_LOAD_TILES(K0)                                                    \
  {                                                                            \
    const float4* src = reinterpret_cast<const float4*>(                       \
        &A[(size_t)(row0 + tr) * K + (K0) + tcb]);                             \
    aReg[0] = src[0]; aReg[1] = src[1]; aReg[2] = src[2]; aReg[3] = src[3];    \
    _Pragma("unroll")                                                          \
    for (int i = 0; i < 16; i++) {                                             \
      int kk = wkb + (i >> 3);                                                 \
      int nn = wnb + (i & 7);                                                  \
      int col = col0 + nn;                                                     \
      int colc = (col < M) ? col : (M - 1);                                    \
      float vv = W[(size_t)((K0) + kk) * M + colc];                            \
      wReg[i] = (col < M) ? vv : 0.f;                                          \
    }                                                                          \
  }

  GEMM_LOAD_TILES(0);

  for (int k0 = 0; k0 < K; k0 += 32) {
    __syncthreads();  // previous compute done reading LDS
    // stage registers -> LDS (f32 -> f16)
#pragma unroll
    for (int i = 0; i < 4; i++) {
      float4 f = aReg[i];
      int c = tcb + i * 4;
      As[tr * AST + c + 0] = (_Float16)f.x;
      As[tr * AST + c + 1] = (_Float16)f.y;
      As[tr * AST + c + 2] = (_Float16)f.z;
      As[tr * AST + c + 3] = (_Float16)f.w;
    }
#pragma unroll
    for (int i = 0; i < 16; i++) {
      int kk = wkb + (i >> 3);
      int nn = wnb + (i & 7);
      Ws[nn * WST + kk] = (_Float16)wReg[i];
    }
    __syncthreads();

    // kick off next tile's global loads; they overlap the WMMA stream below
    if (k0 + 32 < K) GEMM_LOAD_TILES(k0 + 32);

    HF a[2], bfr[4];
#pragma unroll
    for (int rt = 0; rt < 2; rt++) {
      int r = wr * 32 + rt * 16 + l15;
      const _Float16* ap = &As[r * AST];
#pragma unroll
      for (int v = 0; v < 8; v++) {
        int c = hlf * 16 + 2 * v;
        a[rt].u[v] = *reinterpret_cast<const unsigned int*>(ap + c);
      }
    }
#pragma unroll
    for (int ct = 0; ct < 4; ct++) {
      int n = wcg * 64 + ct * 16 + l15;
      const _Float16* bp = &Ws[n * WST + hlf * 16];
#pragma unroll
      for (int v = 0; v < 8; v++)
        bfr[ct].u[v] = *reinterpret_cast<const unsigned int*>(bp + 2 * v);
    }
#pragma unroll
    for (int rt = 0; rt < 2; rt++)
#pragma unroll
      for (int ct = 0; ct < 4; ct++)
        acc[rt][ct] = __builtin_amdgcn_wmma_f32_16x16x32_f16(
            false, a[rt].v, false, bfr[ct].v, (short)0, acc[rt][ct], false, false);
  }
#undef GEMM_LOAD_TILES

#pragma unroll
  for (int rt = 0; rt < 2; rt++) {
#pragma unroll
    for (int ct = 0; ct < 4; ct++) {
      int col = col0 + wcg * 64 + ct * 16 + l15;
      if (col >= M) continue;
#pragma unroll
      for (int g = 0; g < 8; g++) {
        int row = row0 + wr * 32 + rt * 16 + hlf * 8 + g;
        float v = acc[rt][ct][g] + bias[col];
        if (act == 1) v = 0.5f * v * (1.0f + erff(v * 0.70710678118f));
        if (res) v += res[(size_t)row * M + col];
        C[(size_t)row * M + col] = v;
      }
    }
  }
}

// ---------------------------------------------------------------------------
// Flash attention: one workgroup = 8 waves = 8 q-tiles of 16 rows; 2 workgroups
// cover one (b,h). Keys chunked by 32; K/V tiles staged cooperatively.
// Branchless (clamped) staging loads. ps(key)=exp(-scal*(pos-0.5)^2).
// ---------------------------------------------------------------------------
__global__ __launch_bounds__(256) void attn_kernel(
    const float* __restrict__ qkv, float* __restrict__ ao,
    const float* __restrict__ scal_p) {
  constexpr int QST = 98, VST = 34, PST = 34;
  __shared__ __align__(16) _Float16 Qs[8][16 * QST];
  __shared__ __align__(16) _Float16 Ks[32 * QST];
  __shared__ __align__(16) _Float16 Vt[80 * VST];
  __shared__ __align__(16) _Float16 Ps[8][16 * PST];

  const int bh = blockIdx.x;
  const int b = bh >> 3;
  const int h = bh & 7;
  const int tid = threadIdx.x;
  const int wv = tid >> 5;
  const int lane = tid & 31;
  const int l15 = lane & 15;
  const int hlf = lane >> 4;
  const int q0 = (blockIdx.y * 8 + wv) * 16;
  const float scal = scal_p[0];
  const size_t base = (size_t)b * 243 * 1632;

  // Stage Q tile (zero rows >= 243 / channels >= 68; K padded to 96)
  for (int e = lane; e < 16 * 96; e += 32) {
    int r = e / 96, c = e - r * 96;
    int f = q0 + r;
    int fc = (f < 242) ? f : 242;
    int cc = (c < 67) ? c : 67;
    float v = qkv[base + (size_t)fc * 1632 + h * 68 + cc];
    Qs[wv][r * QST + c] = (_Float16)((c < 68 && f < 243) ? v : 0.f);
  }
  HF aQ[3];
#pragma unroll
  for (int s = 0; s < 3; s++)
#pragma unroll
    for (int v = 0; v < 8; v++) {
      int c = s * 32 + hlf * 16 + 2 * v;
      aQ[s].u[v] = *reinterpret_cast<const unsigned int*>(&Qs[wv][l15 * QST + c]);
    }

  float mrow[8], lrow[8];
  v8f o[5];
#pragma unroll
  for (int g = 0; g < 8; g++) { mrow[g] = -1e30f; lrow[g] = 0.f; }
#pragma unroll
  for (int nt = 0; nt < 5; nt++) {
    v8f z = {0.f, 0.f, 0.f, 0.f, 0.f, 0.f, 0.f, 0.f};
    o[nt] = z;
  }

  for (int kc = 0; kc < 8; kc++) {
    __syncthreads();  // previous chunk compute done before overwriting Ks/Vt
    for (int e = tid; e < 32 * 96; e += 256) {
      int kk = e / 96, c = e - kk * 96;
      int key = kc * 32 + kk;
      int keyc = (key < 242) ? key : 242;
      int cc = (c < 67) ? c : 67;
      float v = qkv[base + (size_t)keyc * 1632 + 544 + h * 68 + cc];
      Ks[kk * QST + c] = (_Float16)((c < 68 && key < 243) ? v : 0.f);
    }
    for (int e = tid; e < 32 * 80; e += 256) {
      int key = e / 80, c = e - key * 80;
      int kg = kc * 32 + key;
      int kgc = (kg < 242) ? kg : 242;
      int cc = (c < 67) ? c : 67;
      float v = qkv[base + (size_t)kgc * 1632 + 1088 + h * 68 + cc];
      Vt[c * VST + key] = (_Float16)((c < 68 && kg < 243) ? v : 0.f);
    }
    __syncthreads();

    float P[2][8];
    float mch[8];
#pragma unroll
    for (int g = 0; g < 8; g++) mch[g] = -1e30f;
#pragma unroll
    for (int ct = 0; ct < 2; ct++) {
      v8f s = {0.f, 0.f, 0.f, 0.f, 0.f, 0.f, 0.f, 0.f};
#pragma unroll
      for (int st = 0; st < 3; st++) {
        HF bK;
        const _Float16* kp = &Ks[(ct * 16 + l15) * QST + st * 32 + hlf * 16];
#pragma unroll
        for (int v = 0; v < 8; v++)
          bK.u[v] = *reinterpret_cast<const unsigned int*>(kp + 2 * v);
        s = __builtin_amdgcn_wmma_f32_16x16x32_f16(false, aQ[st].v, false, bK.v,
                                                   (short)0, s, false, false);
      }
      int key = kc * 32 + ct * 16 + l15;
      float pos = (float)key * (1.f / 242.f) - 0.5f;
      float colscale = 0.121267813f * __expf(-scal * pos * pos);
      bool valid = key < 243;
#pragma unroll
      for (int g = 0; g < 8; g++) {
        float sv = valid ? s[g] * colscale : -1e30f;
        P[ct][g] = sv;
        mch[g] = fmaxf(mch[g], sv);
      }
    }
#pragma unroll
    for (int g = 0; g < 8; g++) mch[g] = half_max16(mch[g]);
    float sf[8], rsum[8];
#pragma unroll
    for (int g = 0; g < 8; g++) {
      float mn = fmaxf(mrow[g], mch[g]);
      sf[g] = __expf(mrow[g] - mn);
      mrow[g] = mn;
      rsum[g] = 0.f;
    }
#pragma unroll
    for (int ct = 0; ct < 2; ct++)
#pragma unroll
      for (int g = 0; g < 8; g++) {
        float p = __expf(P[ct][g] - mrow[g]);
        P[ct][g] = p;
        rsum[g] += p;
      }
#pragma unroll
    for (int g = 0; g < 8; g++) {
      rsum[g] = half_sum16(rsum[g]);
      lrow[g] = lrow[g] * sf[g] + rsum[g];
    }
#pragma unroll
    for (int nt = 0; nt < 5; nt++)
#pragma unroll
      for (int g = 0; g < 8; g++) o[nt][g] *= sf[g];

    // Repack P (C-layout) -> A-fragment via per-wave LDS
#pragma unroll
    for (int ct = 0; ct < 2; ct++)
#pragma unroll
      for (int g = 0; g < 8; g++)
        Ps[wv][(hlf * 8 + g) * PST + ct * 16 + l15] = (_Float16)P[ct][g];
    HF aP;
#pragma unroll
    for (int v = 0; v < 8; v++) {
      int c = hlf * 16 + 2 * v;
      aP.u[v] = *reinterpret_cast<const unsigned int*>(&Ps[wv][l15 * PST + c]);
    }
#pragma unroll
    for (int nt = 0; nt < 5; nt++) {
      HF bV;
      const _Float16* vp = &Vt[(nt * 16 + l15) * VST + hlf * 16];
#pragma unroll
      for (int v = 0; v < 8; v++)
        bV.u[v] = *reinterpret_cast<const unsigned int*>(vp + 2 * v);
      o[nt] = __builtin_amdgcn_wmma_f32_16x16x32_f16(false, aP.v, false, bV.v,
                                                     (short)0, o[nt], false, false);
    }
  }

#pragma unroll
  for (int nt = 0; nt < 5; nt++) {
    int cch = nt * 16 + l15;
    if (cch >= 68) continue;
#pragma unroll
    for (int g = 0; g < 8; g++) {
      int q = q0 + hlf * 8 + g;
      if (q < 243)
        ao[((size_t)b * 243 + q) * 544 + h * 68 + cch] = o[nt][g] / lrow[g];
    }
  }
}

// ---------------------------------------------------------------------------
// Adjacency softmax: A[n,i,:] = softmax(mask ? e : -9e15)
// ---------------------------------------------------------------------------
__global__ void adj_kernel(const unsigned char* __restrict__ mask,
                           const float* __restrict__ e, float* __restrict__ A) {
  int t = threadIdx.x;
  if (t >= 68) return;
  int n = t / 17, i = t - n * 17;
  float r[17];
  float mx = -1e30f;
  for (int j = 0; j < 17; j++) {
    float v = mask[n * 289 + i * 17 + j] ? e[n * 289 + i * 17 + j] : -9e15f;
    r[j] = v;
    mx = fmaxf(mx, v);
  }
  float sm = 0.f;
  for (int j = 0; j < 17; j++) { r[j] = __expf(r[j] - mx); sm += r[j]; }
  for (int j = 0; j < 17; j++) A[n * 289 + i * 17 + j] = r[j] / sm;
}

// ---------------------------------------------------------------------------
// Graph conv + joint attention + LN + relu + pos_embed. One wave per token.
// Lane = output channel o in [0,32).
// ---------------------------------------------------------------------------
__global__ __launch_bounds__(32) void gc_kernel(
    const float* __restrict__ x, const float* __restrict__ Adj,
    const float* __restrict__ W, const float* __restrict__ bias,
    const float* __restrict__ qw, const float* __restrict__ qb,
    const float* __restrict__ kw, const float* __restrict__ kb,
    const float* __restrict__ aw, const float* __restrict__ ab,
    const float* __restrict__ lng, const float* __restrict__ lnb,
    const float* __restrict__ pe, float* __restrict__ xe) {
  const int token = blockIdx.x;
  const int lane = threadIdx.x;
  const int f = token % 243;
  __shared__ float xf[34];
  __shared__ float outsS[17][4][33];
  __shared__ float sS[17][4];
  __shared__ float smx[4], ssum[4];

  xf[lane] = x[(size_t)token * 34 + lane];
  if (lane < 2) xf[32 + lane] = x[(size_t)token * 34 + 32 + lane];
  __syncthreads();

  float outs[4][17];
#pragma unroll
  for (int n = 0; n < 4; n++) {
    float hn[17];
    float w0 = W[(n * 2 + 0) * 32 + lane];
    float w1 = W[(n * 2 + 1) * 32 + lane];
#pragma unroll
    for (int j = 0; j < 17; j++) hn[j] = xf[2 * j] * w0 + xf[2 * j + 1] * w1;
#pragma unroll
    for (int i = 0; i < 17; i++) {
      float a = 0.f;
#pragma unroll
      for (int j = 0; j < 17; j++) a += Adj[n * 289 + i * 17 + j] * hn[j];
      outs[n][i] = a;
      outsS[i][n][lane] = a;
    }
  }
  __syncthreads();
  for (int i = 0; i < 17; i++)
    for (int n = 0; n < 4; n++) {
      float qv = qb[lane], kv = kb[lane];
      for (int c = 0; c < 32; c++) {
        float ov = outsS[i][n][c];
        qv += ov * qw[c * 32 + lane];
        kv += ov * kw[c * 32 + lane];
      }
      float t = wave_sum32(tanhf(qv + kv) * aw[lane]);
      if (lane == 0) sS[i][n] = t + ab[0];
    }
  __syncthreads();
  if (lane < 4) {
    float mx = -1e30f;
    for (int i = 0; i < 17; i++) mx = fmaxf(mx, sS[i][lane]);
    float sm = 0.f;
    for (int i = 0; i < 17; i++) sm += __expf(sS[i][lane] - mx);
    smx[lane] = mx;
    ssum[lane] = sm;
  }
  __syncthreads();
  for (int i = 0; i < 17; i++) {
    float att = bias[lane];
#pragma unroll
    for (int n = 0; n < 4; n++)
      att += outs[n][i] * (__expf(sS[i][n] - smx[n]) / ssum[n]);
    float m = wave_sum32(att) * (1.f / 32.f);
    float d = att - m;
    float v = wave_sum32(d * d) * (1.f / 32.f);
    float g = d * rsqrtf(v + 1e-5f) * lng[lane] + lnb[lane];
    g = fmaxf(g, 0.f);
    xe[(size_t)token * 544 + i * 32 + lane] = g + pe[f * 544 + i * 32 + lane];
  }
}

// ---------------------------------------------------------------------------
// Fused: xn=LN_E(xe) (eps 1e-6); per-joint LN_32 (eps 1e-5) -> xr; s=xr@pw+pb
// Lane mapping: channel c = j*32 + lane, so joint j = iteration index.
// ---------------------------------------------------------------------------
__global__ __launch_bounds__(256) void prelude_kernel(
    const float* __restrict__ xe, const float* __restrict__ g1,
    const float* __restrict__ b1, const float* __restrict__ ga,
    const float* __restrict__ ba, const float* __restrict__ pw,
    const float* __restrict__ pb, float* __restrict__ xr,
    float* __restrict__ sj) {
  const int row = blockIdx.x * 8 + (threadIdx.x >> 5);
  const int lane = threadIdx.x & 31;
  float xv[17];
  float s = 0.f;
#pragma unroll
  for (int j = 0; j < 17; j++) {
    xv[j] = xe[(size_t)row * 544 + j * 32 + lane];
    s += xv[j];
  }
  s = wave_sum32(s);
  float m = s * (1.f / 544.f);
  float vs = 0.f;
#pragma unroll
  for (int j = 0; j < 17; j++) { float d = xv[j] - m; vs += d * d; }
  vs = wave_sum32(vs);
  float rs = rsqrtf(vs * (1.f / 544.f) + 1e-6f);
  for (int j = 0; j < 17; j++) {
    float xn = (xv[j] - m) * rs * g1[j * 32 + lane] + b1[j * 32 + lane];
    float mj = wave_sum32(xn) * (1.f / 32.f);
    float d = xn - mj;
    float vj = wave_sum32(d * d) * (1.f / 32.f);
    float xrv = d * rsqrtf(vj + 1e-5f) * ga[lane] + ba[lane];
    xr[(size_t)row * 544 + j * 32 + lane] = xrv;
    float sc = wave_sum32(xrv * pw[lane]);
    if (lane == 0) sj[(size_t)row * 17 + j] = sc + pb[0];
  }
}

// Softmax over frames (F=243) for each (b, j); in-place on s[[B*F],17].
__global__ __launch_bounds__(256) void frame_softmax_kernel(float* __restrict__ s) {
  const int bj = blockIdx.x;
  const int b = bj / 17, j = bj - b * 17;
  const int tid = threadIdx.x;
  const int lane = tid & 31, wv = tid >> 5;
  __shared__ float redm[8], reds[8];
  size_t idx = ((size_t)b * 243 + tid) * 17 + j;
  float v = (tid < 243) ? s[idx] : -1e30f;
  float mx = v;
#pragma unroll
  for (int m = 16; m >= 1; m >>= 1) mx = fmaxf(mx, __shfl_xor(mx, m, 32));
  if (lane == 0) redm[wv] = mx;
  __syncthreads();
  mx = redm[0];
#pragma unroll
  for (int i = 1; i < 8; i++) mx = fmaxf(mx, redm[i]);
  float e = (tid < 243) ? __expf(v - mx) : 0.f;
  float sm = e;
#pragma unroll
  for (int m = 16; m >= 1; m >>= 1) sm += __shfl_xor(sm, m, 32);
  if (lane == 0) reds[wv] = sm;
  __syncthreads();
  sm = 0.f;
#pragma unroll
  for (int i = 0; i < 8; i++) sm += reds[i];
  if (tid < 243) s[idx] = e / sm;
}

// xa = LN_E(xr * wf) with at_norm2 (eps 1e-5)
__global__ __launch_bounds__(256) void pool_ln_kernel(
    const float* __restrict__ xr, const float* __restrict__ wf,
    const float* __restrict__ g, const float* __restrict__ bb,
    float* __restrict__ xa) {
  const int row = blockIdx.x * 8 + (threadIdx.x >> 5);
  const int lane = threadIdx.x & 31;
  float vals[17];
  float s = 0.f;
#pragma unroll
  for (int j = 0; j < 17; j++) {
    float v = xr[(size_t)row * 544 + j * 32 + lane] * wf[(size_t)row * 17 + j];
    vals[j] = v;
    s += v;
  }
  s = wave_sum32(s);
  float m = s * (1.f / 544.f);
  float vs = 0.f;
#pragma unroll
  for (int j = 0; j < 17; j++) { float d = vals[j] - m; vs += d * d; }
  vs = wave_sum32(vs);
  float rs = rsqrtf(vs * (1.f / 544.f) + 1e-5f);
#pragma unroll
  for (int j = 0; j < 17; j++)
    xa[(size_t)row * 544 + j * 32 + lane] =
        (vals[j] - m) * rs * g[j * 32 + lane] + bb[j * 32 + lane];
}

// Generic LayerNorm over E=544 (one wave per row)
__global__ __launch_bounds__(256) void ln_kernel(
    const float* __restrict__ x, const float* __restrict__ g,
    const float* __restrict__ bb, float* __restrict__ y, float eps) {
  const int row = blockIdx.x * 8 + (threadIdx.x >> 5);
  const int lane = threadIdx.x & 31;
  float vals[17];
  float s = 0.f;
#pragma unroll
  for (int j = 0; j < 17; j++) {
    vals[j] = x[(size_t)row * 544 + j * 32 + lane];
    s += vals[j];
  }
  s = wave_sum32(s);
  float m = s * (1.f / 544.f);
  float vs = 0.f;
#pragma unroll
  for (int j = 0; j < 17; j++) { float d = vals[j] - m; vs += d * d; }
  vs = wave_sum32(vs);
  float rs = rsqrtf(vs * (1.f / 544.f) + eps);
#pragma unroll
  for (int j = 0; j < 17; j++)
    y[(size_t)row * 544 + j * 32 + lane] =
        (vals[j] - m) * rs * g[j * 32 + lane] + bb[j * 32 + lane];
}

// ---------------------------------------------------------------------------
extern "C" void kernel_launch(void* const* d_in, const int* in_sizes, int n_in,
                              void* d_out, int out_size, void* d_ws, size_t ws_size,
                              hipStream_t stream) {
  (void)in_sizes; (void)n_in; (void)out_size; (void)ws_size;
  const float* x       = (const float*)d_in[0];
  const unsigned char* adj = (const unsigned char*)d_in[1];
  const float* gc_W    = (const float*)d_in[2];
  const float* gc_e    = (const float*)d_in[3];
  const float* gc_bias = (const float*)d_in[4];
  const float* gc_qw   = (const float*)d_in[5];
  const float* gc_qb   = (const float*)d_in[6];
  const float* gc_kw   = (const float*)d_in[7];
  const float* gc_kb   = (const float*)d_in[8];
  const float* gc_aw   = (const float*)d_in[9];
  const float* gc_ab   = (const float*)d_in[10];
  const float* gc_lng  = (const float*)d_in[11];
  const float* gc_lnb  = (const float*)d_in[12];
  const float* pos     = (const float*)d_in[13];
  const float* n1g     = (const float*)d_in[14];
  const float* n1b     = (const float*)d_in[15];
  const float* atg     = (const float*)d_in[16];
  const float* atb     = (const float*)d_in[17];
  const float* apw     = (const float*)d_in[18];
  const float* apb     = (const float*)d_in[19];
  const float* a2g     = (const float*)d_in[20];
  const float* a2b     = (const float*)d_in[21];
  const float* ascal   = (const float*)d_in[22];
  const float* qkvw    = (const float*)d_in[23];
  const float* qkvb    = (const float*)d_in[24];
  const float* projw   = (const float*)d_in[25];
  const float* projb   = (const float*)d_in[26];
  const float* n2g     = (const float*)d_in[27];
  const float* n2b     = (const float*)d_in[28];
  const float* f1w     = (const float*)d_in[29];
  const float* f1b     = (const float*)d_in[30];
  const float* f2w     = (const float*)d_in[31];
  const float* f2b     = (const float*)d_in[32];
  const float* fng     = (const float*)d_in[33];
  const float* fnb     = (const float*)d_in[34];
  const float* hw      = (const float*)d_in[35];
  const float* hb      = (const float*)d_in[36];

  const size_t N = 31104;  // B*F = 128*243
  float* ws   = (float*)d_ws;
  float* Abuf = ws;                     // 4*17*17 (padded to 2048)
  float* xe   = ws + 2048;              // [N,544]
  float* xr   = xe + N * 544;           // [N,544]
  float* xa   = xr + N * 544;           // [N,544]
  float* wf   = xa + N * 544;           // [N,17] (slot N*32)
  float* qkv  = wf + N * 32;            // [N,1632]
  float* ao   = qkv + N * 1632;         // [N,544]
  float* hm   = qkv;                    // [N,1088] aliases dead qkv

  adj_kernel<<<1, 128, 0, stream>>>(adj, gc_e, Abuf);
  gc_kernel<<<(unsigned)N, 32, 0, stream>>>(x, Abuf, gc_W, gc_bias, gc_qw, gc_qb,
                                            gc_kw, gc_kb, gc_aw, gc_ab, gc_lng,
                                            gc_lnb, pos, xe);

  for (int d = 0; d < 4; d++) {
    prelude_kernel<<<N / 8, 256, 0, stream>>>(xe, n1g + d * 544, n1b + d * 544,
                                              atg + d * 32, atb + d * 32,
                                              apw + d * 32, apb + d, xr, wf);
    frame_softmax_kernel<<<128 * 17, 256, 0, stream>>>(wf);
    pool_ln_kernel<<<N / 8, 256, 0, stream>>>(xr, wf, a2g + d * 544, a2b + d * 544, xa);
    gemm_wmma_kernel<<<dim3(13, 243), 256, 0, stream>>>(
        xa, qkvw + (size_t)d * 544 * 1632, qkvb + d * 1632, nullptr, qkv, 544, 1632, 0);
    attn_kernel<<<dim3(1024, 2), 256, 0, stream>>>(qkv, ao, ascal + d);
    gemm_wmma_kernel<<<dim3(5, 243), 256, 0, stream>>>(
        ao, projw + (size_t)d * 544 * 544, projb + d * 544, xe, xe, 544, 544, 0);
    ln_kernel<<<N / 8, 256, 0, stream>>>(xe, n2g + d * 544, n2b + d * 544, xa, 1e-6f);
    gemm_wmma_kernel<<<dim3(9, 243), 256, 0, stream>>>(
        xa, f1w + (size_t)d * 544 * 1088, f1b + d * 1088, nullptr, hm, 544, 1088, 1);
    gemm_wmma_kernel<<<dim3(5, 243), 256, 0, stream>>>(
        hm, f2w + (size_t)d * 1088 * 544, f2b + d * 544, xe, xe, 1088, 544, 0);
  }
  ln_kernel<<<N / 8, 256, 0, stream>>>(xe, fng, fnb, xa, 1e-6f);
  gemm_wmma_kernel<<<dim3(1, 243), 256, 0, stream>>>(xa, hw, hb, nullptr,
                                                     (float*)d_out, 544, 51, 0);
}
